// LSTM_52939766890605
// MI455X (gfx1250) — compile-verified, matrix operand order checked by reading, not run on hardware
//
#include <hip/hip_runtime.h>

// LSTM: S=512, B=128, IN=256, H=512 on MI455X (gfx1250).
//   g = [x_t | h_{t-1}] (B x 768, bf16) @ [W;U]^T (768 x 2048, bf16) -> f32 (WMMA)
// - Weights packed once to bf16 [4H x 768] (3 MB, L2-resident: the recurrence
//   never touches HBM after the first pass).
// - x pre-converted to bf16 once; h kept as a bf16 ping-pong pair, so per-step
//   A-panel staging is a pure bf16 tile move by the Tensor Data Mover
//   (tensor_load_to_lds + s_wait_tensorcnt) with D# row padding for
//   bank-conflict-free LDS strides.
// - PERSISTENT kernel: 32 co-resident blocks iterate all 512 timesteps with a
//   device-scope atomic barrier per step (pre-zeroed counter per step).
// - A laundered *integer offset* (not a laundered pointer: that demoted the
//   loads to FLAT) blocks LICM from hoisting 96 B-fragments across the
//   timestep loop while keeping GLOBAL_LOAD addressing.

#define S_LEN 512
#define B_SZ 128
#define IN_SZ 256
#define H_SZ 512
#define K_SZ (IN_SZ + H_SZ)   // 768
#define G4H (4 * H_SZ)        // 2048
#define NBLK 32               // persistent grid: (8 btiles) x (4 jgroups)

// padded LDS row strides (halfs): +8 halfs (16B) per row -> conflict-free
#define AX_STRIDE (IN_SZ + 8)  // 264 halfs = 528 B
#define AH_STRIDE (H_SZ + 8)   // 520 halfs = 1040 B

typedef __attribute__((ext_vector_type(16))) __bf16 v16bf;
typedef __attribute__((ext_vector_type(8)))  __bf16 v8bf;
typedef __attribute__((ext_vector_type(4)))  __bf16 v4bf;
typedef __attribute__((ext_vector_type(8)))  float  v8f;
typedef __attribute__((ext_vector_type(4)))  unsigned int v4u;
typedef __attribute__((ext_vector_type(8)))  int v8i;
typedef __attribute__((ext_vector_type(4)))  int v4i;

__device__ __forceinline__ float sig_f(float x) { return 1.0f / (1.0f + __expf(-x)); }
__device__ __forceinline__ float tanh_f(float x) {
    float e = __expf(2.0f * x);
    return (e - 1.0f) / (e + 1.0f);
}

__device__ __forceinline__ v16bf load_frag(const __bf16* p) {
    // 16-bit fragment: two contiguous 8-half (16 B) chunks at K offsets {0, 16}
    v8bf lo = *(const v8bf*)p;
    v8bf hi = *(const v8bf*)(p + 16);
    return __builtin_shufflevector(lo, hi,
        0, 1, 2, 3, 4, 5, 6, 7, 8, 9, 10, 11, 12, 13, 14, 15);
}

// ---------------------------------------------------------------------------
// TDM: 2D bf16 tile (n_rows x n_elem) global -> LDS with row padding.
// D# per CDNA5 ISA 8.3/8.4 (group0: count|lds|global|type=2; group1:
// data_size=2B, pad fields, dims, strides; groups2/3 zero for 2D tiles).
// ---------------------------------------------------------------------------
#if __has_builtin(__builtin_amdgcn_tensor_load_to_lds)
#define HAVE_TDM 1
__device__ __forceinline__ void tdm_load_2d(unsigned int lds_off, const __bf16* gptr,
                                            unsigned int n_elem, unsigned int n_rows,
                                            unsigned int row_stride_elems,
                                            unsigned int pad_interval,
                                            unsigned int pad_amount)
{
    unsigned long long ga = (unsigned long long)(uintptr_t)gptr;
    v4u g0;
    g0.x = 1u;                                                   // count=1
    g0.y = lds_off;                                              // lds_addr (bytes)
    g0.z = (unsigned int)(ga & 0xffffffffu);                     // global_addr[31:0]
    g0.w = (unsigned int)((ga >> 32) & 0x01ffffffu) | (2u << 30);// ga[56:32] | type=2

    v8i g1;
    g1[0] = (int)((1u << 16) |                 // data_size = 2 bytes
                  (1u << 20) |                 // pad_enable
                  (pad_interval << 22) |       // DWORDs-per-row encode
                  (pad_amount << 25));         // pad DWORDs encode
    g1[1] = (int)((n_elem & 0xffffu) << 16);                     // tensor_dim0 lo
    g1[2] = (int)(((n_elem >> 16) & 0xffffu) |
                  ((n_rows & 0xffffu) << 16));                   // td0 hi | tensor_dim1 lo
    g1[3] = (int)(((n_rows >> 16) & 0xffffu) |
                  ((n_elem & 0xffffu) << 16));                   // td1 hi | tile_dim0
    g1[4] = (int)(n_rows & 0xffffu);                             // tile_dim1 | tile_dim2=0
    g1[5] = (int)row_stride_elems;                               // tensor_dim0_stride lo
    g1[6] = 0;
    g1[7] = 0;

    v4i z4 = {0, 0, 0, 0};
#if __clang_major__ >= 23
    v8i z8 = {0, 0, 0, 0, 0, 0, 0, 0};
    __builtin_amdgcn_tensor_load_to_lds(g0, g1, z4, z4, z8, 0);
#else
    __builtin_amdgcn_tensor_load_to_lds(g0, g1, z4, z4, 0);
#endif
}
#else
#define HAVE_TDM 0
#endif

__device__ __forceinline__ void wait_tensorcnt0() {
#if __has_builtin(__builtin_amdgcn_s_wait_tensorcnt)
    __builtin_amdgcn_s_wait_tensorcnt(0);
#elif HAVE_TDM
    asm volatile("s_wait_tensorcnt 0x0" ::: "memory");
#endif
}

// ---------------------------------------------------------------------------
// Pack W{f,i,o,g} [H,IN] and U{f,i,o,g} [H,H] into bf16 Wpack[4H][768],
// bias[u] = bW + bU (row u = gate*H + j).
// ---------------------------------------------------------------------------
__global__ __launch_bounds__(256)
void lstm_pack_kernel(const float* __restrict__ Wf, const float* __restrict__ Wi,
                      const float* __restrict__ Wo, const float* __restrict__ Wg,
                      const float* __restrict__ Uf, const float* __restrict__ Ui,
                      const float* __restrict__ Uo, const float* __restrict__ Ug,
                      const float* __restrict__ bWf, const float* __restrict__ bWi,
                      const float* __restrict__ bWo, const float* __restrict__ bWg,
                      const float* __restrict__ bUf, const float* __restrict__ bUi,
                      const float* __restrict__ bUo, const float* __restrict__ bUg,
                      __bf16* __restrict__ Wpack, float* __restrict__ bias)
{
    int idx = blockIdx.x * 256 + threadIdx.x;
    if (idx >= G4H * K_SZ) return;
    int u = idx / K_SZ;
    int k = idx - u * K_SZ;
    int g = u >> 9;
    int j = u & (H_SZ - 1);

    const float* W[4]  = { Wf, Wi, Wo, Wg };
    const float* U[4]  = { Uf, Ui, Uo, Ug };
    const float* bW[4] = { bWf, bWi, bWo, bWg };
    const float* bU[4] = { bUf, bUi, bUo, bUg };

    float v = (k < IN_SZ) ? W[g][j * IN_SZ + k] : U[g][j * H_SZ + (k - IN_SZ)];
    Wpack[idx] = (__bf16)v;
    if (k == 0) bias[u] = bW[g][j] + bU[g][j];
}

// x (f32) -> xbf (bf16), all timesteps in one parallel pass.
__global__ __launch_bounds__(256)
void lstm_xconv_kernel(const float* __restrict__ x, __bf16* __restrict__ xbf)
{
    size_t i = ((size_t)blockIdx.x * 256 + threadIdx.x) * 4;
    if (i >= (size_t)S_LEN * B_SZ * IN_SZ) return;
    float4 v = *(const float4*)(x + i);
    v4bf w = { (__bf16)v.x, (__bf16)v.y, (__bf16)v.z, (__bf16)v.w };
    *(v4bf*)(xbf + i) = w;
}

__global__ __launch_bounds__(256)
void lstm_init_state(const float* __restrict__ h0, const float* __restrict__ c0,
                     __bf16* __restrict__ hbf, float* __restrict__ cbuf,
                     unsigned int* __restrict__ counters)
{
    int i = blockIdx.x * 256 + threadIdx.x;
    if (i < B_SZ * H_SZ) {
        cbuf[i] = c0[i];
        hbf[i]  = (__bf16)h0[i];
    }
    if (i < S_LEN) counters[i] = 0;   // per-step grid-barrier counters
}

// ---------------------------------------------------------------------------
// Persistent LSTM kernel. grid = (8, 4) = 32 blocks, 256 threads (8 waves).
// Wave w of block (bt, jg) owns the 16x16 hidden tile (bt, jg*8+w) with 4 WMMA
// accumulators (f/i/o/g), so the gate fusion is entirely lane-local.
// ---------------------------------------------------------------------------
__global__ __launch_bounds__(256)
void lstm_persistent_kernel(const __bf16* __restrict__ xbf,    // [S, B, IN]
                            const __bf16* __restrict__ Wpack,  // [4H, K]
                            const float*  __restrict__ bias,   // [4H]
                            float*  __restrict__ cbuf,         // [B, H]
                            __bf16* __restrict__ hbfA,         // [B, H] ping
                            __bf16* __restrict__ hbfB,         // [B, H] pong
                            float*  __restrict__ hseq,         // [S+2, B, H] out
                            unsigned int* __restrict__ counters)
{
    __shared__ __bf16 Ax[16 * AX_STRIDE];
    __shared__ __bf16 Ah[16 * AH_STRIDE];

    const int tid  = threadIdx.x;
    const int b0   = blockIdx.x * 16;
    const int wave = tid >> 5;
    const int lane = tid & 31;
    const int j0   = (blockIdx.y * 8 + wave) * 16;

    // 16-bit fragment lane mapping: lane<16 -> row=lane, K chunks {0..7,16..23};
    // lane>=16 -> row=lane-16, K chunks {8..15,24..31}.
    const int nl = lane & 15;
    const int ol = (lane >> 4) * 8;
    const int rbase = (lane >> 4) * 8;

    const __bf16* abase_x = Ax + nl * AX_STRIDE + ol;
    const __bf16* abase_h = Ah + nl * AH_STRIDE + ol;

    const float bf = bias[0 * H_SZ + j0 + nl];
    const float bi = bias[1 * H_SZ + j0 + nl];
    const float bo = bias[2 * H_SZ + j0 + nl];
    const float bg = bias[3 * H_SZ + j0 + nl];

#pragma unroll 1
    for (int s = 0; s < S_LEN; ++s) {
        const __bf16* xbf_t = xbf + (size_t)s * B_SZ * IN_SZ;
        const __bf16* hprev = (s & 1) ? hbfB : hbfA;
        __bf16*       hnext = (s & 1) ? hbfA : hbfB;
        float*        hout  = hseq + (size_t)s * B_SZ * H_SZ;

        // Launder an integer offset each step: the weight-fragment addresses
        // depend on a value redefined by opaque asm every iteration, so LICM
        // cannot hoist the 96 loop-invariant B-fragment loads across the
        // 512-step loop (which caused ~3 KB/lane scratch spills). Using an
        // integer (not a laundered pointer) preserves addrspace inference,
        // keeping these GLOBAL_LOADs instead of FLAT.
        unsigned int zoff = 0;
        asm volatile("" : "+s"(zoff));
        const __bf16* wbase = Wpack + zoff;
        const __bf16* wrow0 = wbase + (size_t)(0 * H_SZ + j0 + nl) * K_SZ + ol;
        const __bf16* wrow1 = wbase + (size_t)(1 * H_SZ + j0 + nl) * K_SZ + ol;
        const __bf16* wrow2 = wbase + (size_t)(2 * H_SZ + j0 + nl) * K_SZ + ol;
        const __bf16* wrow3 = wbase + (size_t)(3 * H_SZ + j0 + nl) * K_SZ + ol;

        // ---- A-panel staging via Tensor Data Mover (wave 0 issues DMAs) ----
#if HAVE_TDM
        if (tid < 32) {
            // x panel: 16 x 256 halfs (512 B = 128 DW -> pad_interval=6), +16B pad (enc 3)
            tdm_load_2d((unsigned int)(uintptr_t)Ax, xbf_t + (size_t)b0 * IN_SZ,
                        IN_SZ, 16, IN_SZ, 6u, 3u);
            // h panel: 16 x 512 halfs (1024 B = 256 DW -> pad_interval=7), +16B pad
            tdm_load_2d((unsigned int)(uintptr_t)Ah, hprev + (size_t)b0 * H_SZ,
                        H_SZ, 16, H_SZ, 7u, 3u);
            wait_tensorcnt0();
        }
        __syncthreads();
#else
        for (int q = tid; q < 16 * (IN_SZ / 8); q += 256) {
            int r = q / (IN_SZ / 8), kq = (q % (IN_SZ / 8)) * 8;
            *(v8bf*)(Ax + r * AX_STRIDE + kq) = *(const v8bf*)(xbf_t + (b0 + r) * IN_SZ + kq);
        }
        for (int q = tid; q < 16 * (H_SZ / 8); q += 256) {
            int r = q / (H_SZ / 8), kq = (q % (H_SZ / 8)) * 8;
            *(v8bf*)(Ah + r * AH_STRIDE + kq) = *(const v8bf*)(hprev + (b0 + r) * H_SZ + kq);
        }
        __syncthreads();
#endif

        // ---- software-pipelined K loop: 24 iters of 16x16x32 bf16 WMMA ----
        auto aptr = [&](int it) -> const __bf16* {
            return (it < 8) ? (abase_x + it * 32) : (abase_h + (it - 8) * 32);
        };

        v8f acc0 = {}, acc1 = {}, acc2 = {}, acc3 = {};
        v16bf ac  = load_frag(aptr(0));
        v16bf b0c = load_frag(wrow0);
        v16bf b1c = load_frag(wrow1);
        v16bf b2c = load_frag(wrow2);
        v16bf b3c = load_frag(wrow3);

#pragma unroll 4
        for (int it = 0; it < 23; ++it) {
            const int kn = (it + 1) * 32;
            v16bf an  = load_frag(aptr(it + 1));
            v16bf b0n = load_frag(wrow0 + kn);
            v16bf b1n = load_frag(wrow1 + kn);
            v16bf b2n = load_frag(wrow2 + kn);
            v16bf b3n = load_frag(wrow3 + kn);
            acc0 = __builtin_amdgcn_wmma_f32_16x16x32_bf16(false, ac, false, b0c, (short)0, acc0, false, false);
            acc1 = __builtin_amdgcn_wmma_f32_16x16x32_bf16(false, ac, false, b1c, (short)0, acc1, false, false);
            acc2 = __builtin_amdgcn_wmma_f32_16x16x32_bf16(false, ac, false, b2c, (short)0, acc2, false, false);
            acc3 = __builtin_amdgcn_wmma_f32_16x16x32_bf16(false, ac, false, b3c, (short)0, acc3, false, false);
            ac = an; b0c = b0n; b1c = b1n; b2c = b2n; b3c = b3n;
        }
        acc0 = __builtin_amdgcn_wmma_f32_16x16x32_bf16(false, ac, false, b0c, (short)0, acc0, false, false);
        acc1 = __builtin_amdgcn_wmma_f32_16x16x32_bf16(false, ac, false, b1c, (short)0, acc1, false, false);
        acc2 = __builtin_amdgcn_wmma_f32_16x16x32_bf16(false, ac, false, b2c, (short)0, acc2, false, false);
        acc3 = __builtin_amdgcn_wmma_f32_16x16x32_bf16(false, ac, false, b3c, (short)0, acc3, false, false);

        // ---- fused gate epilogue ----
        // C/D layout: VGPR r -> M = r (lanes 0-15) / r+8 (lanes 16-31), N = lane&15.
#pragma unroll
        for (int r = 0; r < 8; ++r) {
            int b   = b0 + rbase + r;
            int idx = b * H_SZ + j0 + nl;
            float f  = sig_f(acc0[r] + bf);
            float ii = sig_f(acc1[r] + bi);
            float oo = sig_f(acc2[r] + bo);
            float cd = tanh_f(acc3[r] + bg);
            float cn = f * cbuf[idx] + ii * cd;
            cbuf[idx] = cn;
            float hh = oo * tanh_f(cn);
            hout[idx]  = hh;
            hnext[idx] = (__bf16)hh;
            if (s == S_LEN - 1) {
                float* hfin = hseq + (size_t)S_LEN * B_SZ * H_SZ;
                hfin[idx] = hh;                 // final h
                hfin[B_SZ * H_SZ + idx] = cn;   // final c
            }
        }

        // ---- device-scope grid barrier between timesteps ----
        if (s < S_LEN - 1) {
            __threadfence();      // release this wave's h/c stores to L2 (agent scope)
            __syncthreads();      // whole block arrived
            if (tid == 0) {
                __hip_atomic_fetch_add(&counters[s], 1u,
                                       __ATOMIC_ACQ_REL, __HIP_MEMORY_SCOPE_AGENT);
                while (__hip_atomic_load(&counters[s],
                                         __ATOMIC_ACQUIRE, __HIP_MEMORY_SCOPE_AGENT) < NBLK) {
                    __builtin_amdgcn_s_sleep(1);
                }
            }
            __syncthreads();      // release whole block; also guards LDS reuse vs next TDM
        }
    }
}

// ---------------------------------------------------------------------------
extern "C" void kernel_launch(void* const* d_in, const int* in_sizes, int n_in,
                              void* d_out, int out_size, void* d_ws, size_t ws_size,
                              hipStream_t stream) {
    const float* x   = (const float*)d_in[0];
    const float* h0  = (const float*)d_in[1];
    const float* c0  = (const float*)d_in[2];
    const float* Wf  = (const float*)d_in[3];
    const float* bWf = (const float*)d_in[4];
    const float* Uf  = (const float*)d_in[5];
    const float* bUf = (const float*)d_in[6];
    const float* Wi  = (const float*)d_in[7];
    const float* bWi = (const float*)d_in[8];
    const float* Ui  = (const float*)d_in[9];
    const float* bUi = (const float*)d_in[10];
    const float* Wo  = (const float*)d_in[11];
    const float* bWo = (const float*)d_in[12];
    const float* Uo  = (const float*)d_in[13];
    const float* bUo = (const float*)d_in[14];
    const float* Wg  = (const float*)d_in[15];
    const float* bWg = (const float*)d_in[16];
    const float* Ug  = (const float*)d_in[17];
    const float* bUg = (const float*)d_in[18];

    // workspace layout
    char* ws = (char*)d_ws;
    size_t off = 0;
    __bf16* Wpack = (__bf16*)(ws + off); off += (size_t)G4H * K_SZ * sizeof(__bf16); // 3 MB
    float*  bias  = (float*)(ws + off);  off += (size_t)G4H * sizeof(float);         // 8 KB
    float*  cbuf  = (float*)(ws + off);  off += (size_t)B_SZ * H_SZ * sizeof(float); // 256 KB
    __bf16* hbfA  = (__bf16*)(ws + off); off += (size_t)B_SZ * H_SZ * sizeof(__bf16);
    __bf16* hbfB  = (__bf16*)(ws + off); off += (size_t)B_SZ * H_SZ * sizeof(__bf16);
    unsigned int* counters = (unsigned int*)(ws + off); off += (size_t)S_LEN * sizeof(unsigned int);
    off = (off + 255) & ~(size_t)255;
    __bf16* xbf   = (__bf16*)(ws + off); off += (size_t)S_LEN * B_SZ * IN_SZ * sizeof(__bf16); // 33.5 MB

    lstm_pack_kernel<<<(G4H * K_SZ + 255) / 256, 256, 0, stream>>>(
        Wf, Wi, Wo, Wg, Uf, Ui, Uo, Ug,
        bWf, bWi, bWo, bWg, bUf, bUi, bUo, bUg, Wpack, bias);
    lstm_xconv_kernel<<<(int)(((size_t)S_LEN * B_SZ * IN_SZ / 4 + 255) / 256), 256, 0, stream>>>(x, xbf);
    lstm_init_state<<<(B_SZ * H_SZ + 255) / 256, 256, 0, stream>>>(h0, c0, hbfA, cbuf, counters);

    lstm_persistent_kernel<<<dim3(8, 4), 256, 0, stream>>>(
        xbf, Wpack, bias, cbuf, hbfA, hbfB, (float*)d_out, counters);
}
